// ImprovedGCN_13202729468404
// MI455X (gfx1250) — matrix operand order repaired and checked by reading, not compiled, not run on hardware
//
#include <hip/hip_runtime.h>
#include <hip/hip_bf16.h>

#define HDIM 128
#define FIN  11

typedef __attribute__((ext_vector_type(16))) __bf16       v16bf;
typedef __attribute__((ext_vector_type(8)))  float        v8f;
typedef __attribute__((ext_vector_type(4)))  unsigned int v4u;

union FragBF {
  v16bf v;
  v4u   q[2];
};

__device__ __forceinline__ unsigned short f2bf(float f) {
  unsigned int u = __float_as_uint(f);
  unsigned int r = u + 0x7fffu + ((u >> 16) & 1u);   // round-to-nearest-even
  return (unsigned short)(r >> 16);
}

// Native CDNA5 non-returning f32 atomic add at device scope (L2 atomic unit).
// Cross-workgroup accumulation => must be SCOPE_DEV.
__device__ __forceinline__ void atomAddF(float* p, float v) {
  asm volatile("global_atomic_add_f32 %0, %1, off scope:SCOPE_DEV"
               :: "v"((unsigned long long)(uintptr_t)p), "v"(v)
               : "memory");
}

// ---------------- degree / normalization ----------------
__global__ void k_deg_init(float* __restrict__ deg, int N) {
  int i = blockIdx.x * 256 + threadIdx.x;
  if (i < N) deg[i] = 1.0f;                 // self-loop
}

__global__ void k_deg_edges(const int* __restrict__ dst, float* __restrict__ deg, int E) {
  int e = blockIdx.x * 256 + threadIdx.x;
  if (e < E) atomAddF(&deg[dst[e]], 1.0f);
}

__global__ void k_deg_rsqrt(float* __restrict__ deg, int N) {
  int i = blockIdx.x * 256 + threadIdx.x;
  if (i < N) deg[i] = rsqrtf(deg[i]);       // deg becomes dinv in place
}

// ---------------- layer 0 GEMM (K=11, VALU) ----------------
__global__ void k_gemm0(const float* __restrict__ x, const float* __restrict__ W0,
                        float* __restrict__ h, int N) {
  int n = blockIdx.x;
  int c = threadIdx.x;
  if (n >= N) return;
  __shared__ float xr[FIN];
  if (c < FIN) xr[c] = x[(size_t)n * FIN + c];
  __syncthreads();
  float acc = 0.f;
#pragma unroll
  for (int k = 0; k < FIN; ++k) acc += xr[k] * W0[k * HDIM + c];
  h[(size_t)n * HDIM + c] = acc;
}

// ---------------- f32 -> bf16 activation convert ----------------
__global__ void k_convert(const float* __restrict__ x, unsigned short* __restrict__ xbf, int total) {
  int i = blockIdx.x * 256 + threadIdx.x;
  if (i < total) xbf[i] = f2bf(x[i]);
}

// ---------------- pack W (128x128, row-major KxN) into WMMA B-fragment layout ----------------
// B fragment (16-bit, 32x16): lane l holds column n=l&15; half=l>>4; element e -> K = kc*32 + (e<8 ? half*8 + e : 16 + half*8 + (e-8))
__global__ void k_packw(const float* __restrict__ W, unsigned short* __restrict__ Bp) {
  int t = blockIdx.x * 256 + threadIdx.x;   // 0..1023 == nt*128 + kc*32 + lane
  int lane = t & 31;
  int kc   = (t >> 5) & 3;
  int nt   = t >> 7;
  int n    = nt * 16 + (lane & 15);
  int half = lane >> 4;
#pragma unroll
  for (int e = 0; e < 16; ++e) {
    int k = kc * 32 + (e < 8 ? half * 8 + e : 16 + half * 8 + (e - 8));
    Bp[t * 16 + e] = f2bf(W[k * HDIM + n]);
  }
}

// ---------------- WMMA GEMM: h[N x 128] = xbf[N x 128] @ W (pre-packed, staged in LDS) ----------------
__global__ __launch_bounds__(128) void k_gemm_wmma(const unsigned short* __restrict__ Abf,
                                                   const unsigned short* __restrict__ Bp,
                                                   float* __restrict__ H, int N, int mtiles) {
  __shared__ unsigned short Bs[16384];      // 32 KB packed W, shared by 4 waves
  {
    const v4u* gsrc = (const v4u*)Bp;
    v4u* ldst = (v4u*)Bs;
    for (int i = threadIdx.x; i < 2048; i += 128) ldst[i] = gsrc[i];
  }
  __syncthreads();                          // before any wave can exit

  int wave  = threadIdx.x >> 5;
  int lane  = threadIdx.x & 31;
  int mtile = blockIdx.x * 4 + wave;
  if (mtile >= mtiles) return;              // uniform per wave -> EXEC stays all-1s
  int m    = lane & 15;
  int half = lane >> 4;
  int rowA = mtile * 16 + m;
  if (rowA >= N) rowA = N - 1;              // clamp keeps EXEC full
  const unsigned short* arow = Abf + (size_t)rowA * HDIM;

  v8f acc[8] = {};
#pragma unroll
  for (int kc = 0; kc < 4; ++kc) {
    // A fragment: 16-bit 16x32 layout -> two contiguous 8xbf16 runs per lane
    FragBF a;
    int k0 = kc * 32 + half * 8;
    a.q[0] = *(const v4u*)(arow + k0);
    a.q[1] = *(const v4u*)(arow + k0 + 16);
    // load all 8 B fragments for this K chunk first (ds loads overlap),
    // then issue the 8 WMMAs back-to-back
    FragBF b[8];
#pragma unroll
    for (int nt = 0; nt < 8; ++nt)
      b[nt].v = *(const v16bf*)(Bs + ((nt * 4 + kc) * 32 + lane) * 16);
#pragma unroll
    for (int nt = 0; nt < 8; ++nt)
      acc[nt] = __builtin_amdgcn_wmma_f32_16x16x32_bf16(
          false, a.v, false, b[nt].v, (short)0, acc[nt], false, false);
  }
  // C/D layout: lane holds col n=lane&15, rows (half*8 + r)
#pragma unroll
  for (int nt = 0; nt < 8; ++nt) {
#pragma unroll
    for (int r = 0; r < 8; ++r) {
      int row = mtile * 16 + half * 8 + r;
      if (row < N) H[(size_t)row * HDIM + nt * 16 + m] = acc[nt][r];
    }
  }
}

// ---------------- aggregation ----------------
__global__ void k_agg_self(const float* __restrict__ h, const float* __restrict__ dinv,
                           float* __restrict__ agg, int total) {
  int i = blockIdx.x * 256 + threadIdx.x;
  if (i >= total) return;
  int n = i >> 7;
  float d = dinv[n];
  agg[i] = h[i] * d * d;
}

__global__ void k_agg_edges(const float* __restrict__ h, const int* __restrict__ src,
                            const int* __restrict__ dst, const float* __restrict__ dinv,
                            float* __restrict__ agg, int E) {
  int e    = blockIdx.x * 8 + (threadIdx.x >> 5);
  int lane = threadIdx.x & 31;
  if (e >= E) return;
  int s = src[e], d = dst[e];
  float w = dinv[s] * dinv[d];
  const float4* hp = (const float4*)(h + (size_t)s * HDIM);
  float4 v = hp[lane];
  float* ap = agg + (size_t)d * HDIM + lane * 4;
  atomAddF(ap + 0, v.x * w);
  atomAddF(ap + 1, v.y * w);
  atomAddF(ap + 2, v.z * w);
  atomAddF(ap + 3, v.w * w);
}

// ---------------- bias + BN(eval) + ReLU + residual (in place on x) ----------------
__global__ void k_post(const float* __restrict__ agg, const float* __restrict__ bias,
                       const float* __restrict__ gamma, const float* __restrict__ beta,
                       const float* __restrict__ mean, const float* __restrict__ var,
                       float* __restrict__ x, int residual, int total) {
  int i = blockIdx.x * 256 + threadIdx.x;
  if (i >= total) return;
  int c = i & (HDIM - 1);
  float inv = rsqrtf(var[c] + 1e-5f);
  float v = (agg[i] + bias[c] - mean[c]) * inv * gamma[c] + beta[c];
  v = fmaxf(v, 0.f);
  x[i] = residual ? (x[i] + v) : v;
}

// ---------------- global mean pool + MLP ----------------
__global__ void k_pool_zero(float* __restrict__ sums, float* __restrict__ cnts, int G) {
  int i = blockIdx.x * 256 + threadIdx.x;
  if (i < G * HDIM) sums[i] = 0.f;
  if (i < G) cnts[i] = 0.f;
}

__global__ void k_pool_accum(const float* __restrict__ x, const int* __restrict__ batch,
                             float* __restrict__ sums, float* __restrict__ cnts, int N) {
  int n    = blockIdx.x * 8 + (threadIdx.x >> 5);
  int lane = threadIdx.x & 31;
  if (n >= N) return;
  int g = batch[n];
  const float4* xp = (const float4*)(x + (size_t)n * HDIM);
  float4 v = xp[lane];
  float* sp = sums + (size_t)g * HDIM + lane * 4;
  atomAddF(sp + 0, v.x);
  atomAddF(sp + 1, v.y);
  atomAddF(sp + 2, v.z);
  atomAddF(sp + 3, v.w);
  if (lane == 0) atomAddF(&cnts[g], 1.0f);
}

__global__ void k_mlp(const float* __restrict__ sums, const float* __restrict__ cnts,
                      const float* __restrict__ w1, const float* __restrict__ b1,
                      const float* __restrict__ w2, const float* __restrict__ b2,
                      float* __restrict__ out) {
  __shared__ float pooled[HDIM];
  int g = blockIdx.x;
  int lane = threadIdx.x;
  float cnt = fmaxf(cnts[g], 1.0f);
  for (int k = lane; k < HDIM; k += 32) pooled[k] = sums[(size_t)g * HDIM + k] / cnt;
  __syncthreads();
  float a0 = b1[lane];
  float a1 = b1[lane + 32];
  for (int k = 0; k < HDIM; ++k) {
    float p = pooled[k];
    a0 += p * w1[k * 64 + lane];
    a1 += p * w1[k * 64 + lane + 32];
  }
  a0 = fmaxf(a0, 0.f);
  a1 = fmaxf(a1, 0.f);
  float part = a0 * w2[lane] + a1 * w2[lane + 32];
  for (int off = 16; off > 0; off >>= 1) part += __shfl_xor(part, off, 32);
  if (lane == 0) out[g] = part + b2[0];
}

// ---------------- host ----------------
extern "C" void kernel_launch(void* const* d_in, const int* in_sizes, int n_in,
                              void* d_out, int out_size, void* d_ws, size_t ws_size,
                              hipStream_t stream) {
  const float* x_in  = (const float*)d_in[0];
  const int*   ei    = (const int*)d_in[1];
  const int*   batch = (const int*)d_in[2];
  const float* W0    = (const float*)d_in[3];
  const float* b0    = (const float*)d_in[4];
  const float* Ws    = (const float*)d_in[5];
  const float* bs    = (const float*)d_in[6];
  const float* gamma = (const float*)d_in[7];
  const float* beta  = (const float*)d_in[8];
  const float* mean  = (const float*)d_in[9];
  const float* var   = (const float*)d_in[10];
  const float* w1    = (const float*)d_in[11];
  const float* b1    = (const float*)d_in[12];
  const float* w2    = (const float*)d_in[13];
  const float* b2    = (const float*)d_in[14];
  float* out = (float*)d_out;

  const int N = in_sizes[0] / FIN;
  const int E = in_sizes[1] / 2;
  const int G = out_size;
  const int total = N * HDIM;

  char* ws = (char*)d_ws;
  float* x            = (float*)ws;           ws += (size_t)total * 4;
  float* h            = (float*)ws;           ws += (size_t)total * 4;
  float* agg          = (float*)ws;           ws += (size_t)total * 4;
  unsigned short* xbf = (unsigned short*)ws;  ws += (size_t)total * 2;
  unsigned short* wbf = (unsigned short*)ws;  ws += (size_t)16384 * 2;
  float* deg          = (float*)ws;           ws += (size_t)N * 4;   // becomes dinv
  float* sums         = (float*)ws;           ws += (size_t)G * HDIM * 4;
  float* cnts         = (float*)ws;           ws += (size_t)G * 4;

  const int* src = ei;
  const int* dst = ei + E;

  // degrees / symmetric normalization
  k_deg_init <<<(N + 255) / 256, 256, 0, stream>>>(deg, N);
  k_deg_edges<<<(E + 255) / 256, 256, 0, stream>>>(dst, deg, E);
  k_deg_rsqrt<<<(N + 255) / 256, 256, 0, stream>>>(deg, N);

  // layer 0 (K=11, plain VALU GEMM)
  k_gemm0    <<<N, HDIM, 0, stream>>>(x_in, W0, h, N);
  k_agg_self <<<(total + 255) / 256, 256, 0, stream>>>(h, deg, agg, total);
  k_agg_edges<<<(E + 7) / 8, 256, 0, stream>>>(h, src, dst, deg, agg, E);
  k_post     <<<(total + 255) / 256, 256, 0, stream>>>(agg, b0, gamma, beta, mean, var, x, 0, total);

  // layers 1..4 (WMMA bf16 GEMM)
  const int mtiles = (N + 15) / 16;
  for (int li = 1; li < 5; ++li) {
    k_convert  <<<(total + 255) / 256, 256, 0, stream>>>(x, xbf, total);
    k_packw    <<<4, 256, 0, stream>>>(Ws + (size_t)(li - 1) * HDIM * HDIM, wbf);
    k_gemm_wmma<<<(mtiles + 3) / 4, 128, 0, stream>>>(xbf, wbf, h, N, mtiles);
    k_agg_self <<<(total + 255) / 256, 256, 0, stream>>>(h, deg, agg, total);
    k_agg_edges<<<(E + 7) / 8, 256, 0, stream>>>(h, src, dst, deg, agg, E);
    k_post     <<<(total + 255) / 256, 256, 0, stream>>>(agg, bs + (size_t)(li - 1) * HDIM,
                                                         gamma + (size_t)li * HDIM,
                                                         beta + (size_t)li * HDIM,
                                                         mean + (size_t)li * HDIM,
                                                         var + (size_t)li * HDIM, x, 1, total);
  }

  // pooling + MLP head
  k_pool_zero <<<(G * HDIM + 255) / 256, 256, 0, stream>>>(sums, cnts, G);
  k_pool_accum<<<(N + 7) / 8, 256, 0, stream>>>(x, batch, sums, cnts, N);
  k_mlp       <<<G, 32, 0, stream>>>(sums, cnts, w1, b1, w2, b2, out);
}
